// MultiHeadAttention_89120571392727
// MI455X (gfx1250) — matrix operand order repaired and checked by reading, not compiled, not run on hardware
//
#include <hip/hip_runtime.h>
#include <hip/hip_bf16.h>

typedef __attribute__((ext_vector_type(16))) __bf16       v16bf;
typedef __attribute__((ext_vector_type(8)))  float        v8f;
typedef __attribute__((ext_vector_type(8)))  unsigned int u32x8;

#define D_MODEL 1024
#define N_HEADS 16
#define DKV 64
#define BB 4
#define LL 2048
#define NEGV (-1e9f)
#define LN_EPS 1e-5f

#define ASTR 40   // padded K-stride (u16) for A tile in LDS; 80B rows -> 16B aligned chunks
#define BSTR 40   // padded K-stride (u16) for B tile in LDS (stored N-major)

__device__ __forceinline__ unsigned short f2bf(float f) {
  unsigned int u = __builtin_bit_cast(unsigned int, f);
  u += 0x7FFFu + ((u >> 16) & 1u);   // round-to-nearest-even
  return (unsigned short)(u >> 16);
}

// Low 32 bits of a generic shared-aperture address = wave-relative LDS offset.
__device__ __forceinline__ unsigned lds_addr(const void* p) {
  return (unsigned)(unsigned long long)p;
}

// CDNA5 direct global->LDS async copy, 16 bytes per lane (GV mode, ASYNCcnt).
__device__ __forceinline__ void async_copy16(unsigned lds_off, const void* gptr) {
  asm volatile("global_load_async_to_lds_b128 %0, %1, off"
               :: "v"(lds_off), "v"((unsigned long long)gptr)
               : "memory");
}
__device__ __forceinline__ void wait_async() {
  asm volatile("s_wait_asynccnt 0x0" ::: "memory");
}

// A fragment: 16-bit A-matrix 16x32 layout (ISA 7.12.2).
// lane m = lane&15 selects row; element e -> K = (e&7) + 16*(e>>3) + 8*(lane>>4).
__device__ __forceinline__ v16bf ld_a_frag(const unsigned short* As, int row0) {
  int lane = threadIdx.x & 31;
  int h = lane >> 4, m = lane & 15;
  const unsigned short* base = As + (row0 + m) * ASTR;
  u32x8 r;
#pragma unroll
  for (int j = 0; j < 8; ++j) {
    int k0 = ((2 * j) & 7) + ((j & 4) << 2) + (h << 3);
    r[j] = *(const unsigned int*)(base + k0);
  }
  return __builtin_bit_cast(v16bf, r);
}

// B fragment: 32x16 16-bit B. lane n = lane&15 selects column;
// element e -> K = (e&7) + 16*(e>>3) + 8*(lane>>4) (same as A). LDS B is N-major ([n][k]).
__device__ __forceinline__ v16bf ld_b_frag(const unsigned short* Bs, int col0) {
  int lane = threadIdx.x & 31;
  int h = lane >> 4, n = lane & 15;
  const unsigned short* base = Bs + (col0 + n) * BSTR;
  u32x8 r;
#pragma unroll
  for (int j = 0; j < 8; ++j) r[j] = *(const unsigned int*)(base + ((2 * j) & 7) + ((j & 4) << 2) + (h << 3));
  return __builtin_bit_cast(v16bf, r);
}

__device__ __forceinline__ v8f wmma_bf16(v16bf a, v16bf b, v8f c) {
  return __builtin_amdgcn_wmma_f32_16x16x32_bf16(false, a, false, b, (short)0, c,
                                                 false, false);
}

// ---------------------------------------------------------------- cast kernel
__global__ void __launch_bounds__(256)
cast_f32_bf16(const float* __restrict__ src, unsigned short* __restrict__ dst, int n) {
  int i = (blockIdx.x * 256 + threadIdx.x) * 4;
  if (i + 3 < n) {
    float4 v = *(const float4*)(src + i);
    dst[i + 0] = f2bf(v.x);
    dst[i + 1] = f2bf(v.y);
    dst[i + 2] = f2bf(v.z);
    dst[i + 3] = f2bf(v.w);
  }
}

// --------------------------------------------------- projection GEMM (q/k/v)
// out_bf16[8192,1024] = bf16( X(8192x1024) @ W(1024x1024) + bias )
__global__ void __launch_bounds__(256)
proj_gemm(const unsigned short* __restrict__ Xb, const unsigned short* __restrict__ Wb,
          const float* __restrict__ bias, unsigned short* __restrict__ outb) {
  __shared__ __align__(16) unsigned short As[128 * ASTR];
  __shared__ __align__(16) unsigned short Bs[64 * BSTR];
  const int t = threadIdx.x;
  const int wave = t >> 5, wm = wave >> 1, wn = wave & 1;
  const int row0 = blockIdx.x * 128, col0 = blockIdx.y * 64;
  v8f acc[2][2];
#pragma unroll
  for (int a = 0; a < 2; ++a)
#pragma unroll
    for (int b = 0; b < 2; ++b)
#pragma unroll
      for (int e = 0; e < 8; ++e) acc[a][b][e] = 0.f;

  for (int k0 = 0; k0 < 1024; k0 += 32) {
    __syncthreads();
#pragma unroll
    for (int it = 0; it < 2; ++it) {           // A: 128x32 bf16, async global->LDS
      int id = t + 256 * it;
      int r = id >> 2, kg = (id & 3) * 8;
      async_copy16(lds_addr(&As[r * ASTR + kg]),
                   Xb + (size_t)(row0 + r) * 1024 + k0 + kg);
    }
    {                                          // B: 32x64 bf16 -> N-major (transpose)
      int kk = t >> 3, ng = (t & 7) * 8;
      const unsigned short* g = Wb + (size_t)(k0 + kk) * 1024 + col0 + ng;
#pragma unroll
      for (int i = 0; i < 8; ++i) Bs[(ng + i) * BSTR + kk] = g[i];
    }
    wait_async();
    __syncthreads();
    v16bf a0 = ld_a_frag(As, wm * 32);
    v16bf a1 = ld_a_frag(As, wm * 32 + 16);
    v16bf b0 = ld_b_frag(Bs, wn * 32);
    v16bf b1 = ld_b_frag(Bs, wn * 32 + 16);
    acc[0][0] = wmma_bf16(a0, b0, acc[0][0]);
    acc[0][1] = wmma_bf16(a0, b1, acc[0][1]);
    acc[1][0] = wmma_bf16(a1, b0, acc[1][0]);
    acc[1][1] = wmma_bf16(a1, b1, acc[1][1]);
  }
  int lane = t & 31, h = lane >> 4, n = lane & 15;
#pragma unroll
  for (int ti = 0; ti < 2; ++ti)
#pragma unroll
    for (int tj = 0; tj < 2; ++tj)
#pragma unroll
      for (int r = 0; r < 8; ++r) {
        int gr = row0 + wm * 32 + ti * 16 + r + (h << 3);
        int gc = col0 + wn * 32 + tj * 16 + n;
        outb[(size_t)gr * 1024 + gc] = f2bf(acc[ti][tj][r] + bias[gc]);
      }
}

// ------------------------------------------------------ scores = q k^T /8, mask
__global__ void __launch_bounds__(256)
scores_gemm(const unsigned short* __restrict__ qb, const unsigned short* __restrict__ kb,
            const unsigned char* __restrict__ mask, float* __restrict__ attn) {
  __shared__ __align__(16) unsigned short As[128 * ASTR];
  __shared__ __align__(16) unsigned short Bs[64 * BSTR];
  const int t = threadIdx.x;
  const int wave = t >> 5, wm = wave >> 1, wn = wave & 1;
  const int bh = blockIdx.z, b = bh >> 4, hb = (bh & 15) * DKV;
  const int i0 = blockIdx.x * 128, j0 = blockIdx.y * 64;
  v8f acc[2][2];
#pragma unroll
  for (int a = 0; a < 2; ++a)
#pragma unroll
    for (int c = 0; c < 2; ++c)
#pragma unroll
      for (int e = 0; e < 8; ++e) acc[a][c][e] = 0.f;

  for (int d0 = 0; d0 < DKV; d0 += 32) {
    __syncthreads();
#pragma unroll
    for (int it = 0; it < 2; ++it) {            // A: q rows (d contiguous), async
      int id = t + 256 * it;
      int r = id >> 2, kg = (id & 3) * 8;
      async_copy16(lds_addr(&As[r * ASTR + kg]),
                   qb + (size_t)(b * LL + i0 + r) * 1024 + hb + d0 + kg);
    }
    {                                           // B = k^T: row j, d contiguous, async
      int n = t >> 2, dg = (t & 3) * 8;
      async_copy16(lds_addr(&Bs[n * BSTR + dg]),
                   kb + (size_t)(b * LL + j0 + n) * 1024 + hb + d0 + dg);
    }
    wait_async();
    __syncthreads();
    v16bf a0 = ld_a_frag(As, wm * 32);
    v16bf a1 = ld_a_frag(As, wm * 32 + 16);
    v16bf b0 = ld_b_frag(Bs, wn * 32);
    v16bf b1 = ld_b_frag(Bs, wn * 32 + 16);
    acc[0][0] = wmma_bf16(a0, b0, acc[0][0]);
    acc[0][1] = wmma_bf16(a0, b1, acc[0][1]);
    acc[1][0] = wmma_bf16(a1, b0, acc[1][0]);
    acc[1][1] = wmma_bf16(a1, b1, acc[1][1]);
  }
  const float scale = 0.125f;  // 1/sqrt(64)
  int lane = t & 31, h = lane >> 4, n = lane & 15;
#pragma unroll
  for (int ti = 0; ti < 2; ++ti)
#pragma unroll
    for (int tj = 0; tj < 2; ++tj)
#pragma unroll
      for (int r = 0; r < 8; ++r) {
        int i = i0 + wm * 32 + ti * 16 + r + (h << 3);
        int j = j0 + wn * 32 + tj * 16 + n;
        unsigned char mv = mask[((size_t)b * LL + i) * LL + j];
        attn[((size_t)bh * LL + i) * LL + j] = mv ? NEGV : acc[ti][tj][r] * scale;
      }
}

// ------------------------------------------------------------- row softmax
__global__ void __launch_bounds__(256)
softmax_rows(float* __restrict__ attn) {
  __shared__ float red[256];
  float* p = attn + (size_t)blockIdx.x * LL;
  int t = threadIdx.x;
  float v[8], mx = -3.0e38f;
#pragma unroll
  for (int i = 0; i < 8; ++i) { v[i] = p[t + i * 256]; mx = fmaxf(mx, v[i]); }
  red[t] = mx; __syncthreads();
  for (int s = 128; s > 0; s >>= 1) { if (t < s) red[t] = fmaxf(red[t], red[t + s]); __syncthreads(); }
  mx = red[0]; __syncthreads();
  float sum = 0.f;
#pragma unroll
  for (int i = 0; i < 8; ++i) { v[i] = __expf(v[i] - mx); sum += v[i]; }
  red[t] = sum; __syncthreads();
  for (int s = 128; s > 0; s >>= 1) { if (t < s) red[t] += red[t + s]; __syncthreads(); }
  float inv = 1.f / red[0];
#pragma unroll
  for (int i = 0; i < 8; ++i) p[t + i * 256] = v[i] * inv;
}

// ----------------------------------------------------- context = attn @ v
__global__ void __launch_bounds__(256)
context_gemm(const float* __restrict__ attn, const unsigned short* __restrict__ vb,
             unsigned short* __restrict__ ctxb) {
  __shared__ __align__(16) unsigned short As[128 * ASTR];
  __shared__ __align__(16) unsigned short Bs[64 * BSTR];
  const int t = threadIdx.x;
  const int wave = t >> 5, wm = wave >> 1, wn = wave & 1;
  const int bh = blockIdx.z, b = bh >> 4, hb = (bh & 15) * DKV;
  const int i0 = blockIdx.x * 128;
  v8f acc[2][2];
#pragma unroll
  for (int a = 0; a < 2; ++a)
#pragma unroll
    for (int c = 0; c < 2; ++c)
#pragma unroll
      for (int e = 0; e < 8; ++e) acc[a][c][e] = 0.f;

  for (int k0 = 0; k0 < LL; k0 += 32) {
    __syncthreads();
#pragma unroll
    for (int it = 0; it < 2; ++it) {            // A: attn fp32 -> bf16 (convert, manual)
      int id = t + 256 * it;
      int r = id >> 2, kg = (id & 3) * 8;
      const float* g = attn + ((size_t)bh * LL + i0 + r) * LL + k0 + kg;
#pragma unroll
      for (int i = 0; i < 8; ++i) As[r * ASTR + kg + i] = f2bf(g[i]);
    }
    {                                           // B: v rows k, n contiguous (transpose)
      int kk = t >> 3, ng = (t & 7) * 8;
      const unsigned short* g = vb + (size_t)(b * LL + k0 + kk) * 1024 + hb + ng;
#pragma unroll
      for (int i = 0; i < 8; ++i) Bs[(ng + i) * BSTR + kk] = g[i];
    }
    __syncthreads();
    v16bf a0 = ld_a_frag(As, wm * 32);
    v16bf a1 = ld_a_frag(As, wm * 32 + 16);
    v16bf b0 = ld_b_frag(Bs, wn * 32);
    v16bf b1 = ld_b_frag(Bs, wn * 32 + 16);
    acc[0][0] = wmma_bf16(a0, b0, acc[0][0]);
    acc[0][1] = wmma_bf16(a0, b1, acc[0][1]);
    acc[1][0] = wmma_bf16(a1, b0, acc[1][0]);
    acc[1][1] = wmma_bf16(a1, b1, acc[1][1]);
  }
  int lane = t & 31, h = lane >> 4, n = lane & 15;
#pragma unroll
  for (int ti = 0; ti < 2; ++ti)
#pragma unroll
    for (int tj = 0; tj < 2; ++tj)
#pragma unroll
      for (int r = 0; r < 8; ++r) {
        int i = i0 + wm * 32 + ti * 16 + r + (h << 3);
        int c = wn * 32 + tj * 16 + n;  // 0..63 within head
        ctxb[(size_t)(b * LL + i) * 1024 + hb + c] = f2bf(acc[ti][tj][r]);
      }
}

// -------------------------------- out = ctx @ Wo + bo + Q  (pre-layernorm)
__global__ void __launch_bounds__(256)
outproj_gemm(const unsigned short* __restrict__ Cb, const unsigned short* __restrict__ Wb,
             const float* __restrict__ bo, const float* __restrict__ Qin,
             float* __restrict__ y) {
  __shared__ __align__(16) unsigned short As[128 * ASTR];
  __shared__ __align__(16) unsigned short Bs[64 * BSTR];
  const int t = threadIdx.x;
  const int wave = t >> 5, wm = wave >> 1, wn = wave & 1;
  const int row0 = blockIdx.x * 128, col0 = blockIdx.y * 64;
  v8f acc[2][2];
#pragma unroll
  for (int a = 0; a < 2; ++a)
#pragma unroll
    for (int c = 0; c < 2; ++c)
#pragma unroll
      for (int e = 0; e < 8; ++e) acc[a][c][e] = 0.f;

  for (int k0 = 0; k0 < 1024; k0 += 32) {
    __syncthreads();
#pragma unroll
    for (int it = 0; it < 2; ++it) {            // A: ctx bf16, async global->LDS
      int id = t + 256 * it;
      int r = id >> 2, kg = (id & 3) * 8;
      async_copy16(lds_addr(&As[r * ASTR + kg]),
                   Cb + (size_t)(row0 + r) * 1024 + k0 + kg);
    }
    {
      int kk = t >> 3, ng = (t & 7) * 8;
      const unsigned short* g = Wb + (size_t)(k0 + kk) * 1024 + col0 + ng;
#pragma unroll
      for (int i = 0; i < 8; ++i) Bs[(ng + i) * BSTR + kk] = g[i];
    }
    wait_async();
    __syncthreads();
    v16bf a0 = ld_a_frag(As, wm * 32);
    v16bf a1 = ld_a_frag(As, wm * 32 + 16);
    v16bf b0 = ld_b_frag(Bs, wn * 32);
    v16bf b1 = ld_b_frag(Bs, wn * 32 + 16);
    acc[0][0] = wmma_bf16(a0, b0, acc[0][0]);
    acc[0][1] = wmma_bf16(a0, b1, acc[0][1]);
    acc[1][0] = wmma_bf16(a1, b0, acc[1][0]);
    acc[1][1] = wmma_bf16(a1, b1, acc[1][1]);
  }
  int lane = t & 31, h = lane >> 4, n = lane & 15;
#pragma unroll
  for (int ti = 0; ti < 2; ++ti)
#pragma unroll
    for (int tj = 0; tj < 2; ++tj)
#pragma unroll
      for (int r = 0; r < 8; ++r) {
        int gr = row0 + wm * 32 + ti * 16 + r + (h << 3);
        int gc = col0 + wn * 32 + tj * 16 + n;
        y[(size_t)gr * 1024 + gc] = acc[ti][tj][r] + bo[gc] + Qin[(size_t)gr * 1024 + gc];
      }
}

// ------------------------------------------------------------- layer norm
__global__ void __launch_bounds__(256)
layernorm_rows(const float* __restrict__ y, const float* __restrict__ gamma,
               const float* __restrict__ beta, float* __restrict__ out) {
  __shared__ float red[256];
  const float* p = y + (size_t)blockIdx.x * 1024;
  int t = threadIdx.x;
  float v[4], s = 0.f;
#pragma unroll
  for (int i = 0; i < 4; ++i) { v[i] = p[t + i * 256]; s += v[i]; }
  red[t] = s; __syncthreads();
  for (int st = 128; st > 0; st >>= 1) { if (t < st) red[t] += red[t + st]; __syncthreads(); }
  float mu = red[0] * (1.f / 1024.f); __syncthreads();
  s = 0.f;
#pragma unroll
  for (int i = 0; i < 4; ++i) { float d = v[i] - mu; s += d * d; }
  red[t] = s; __syncthreads();
  for (int st = 128; st > 0; st >>= 1) { if (t < st) red[t] += red[t + st]; __syncthreads(); }
  float rstd = rsqrtf(red[0] * (1.f / 1024.f) + LN_EPS);
#pragma unroll
  for (int i = 0; i < 4; ++i) {
    int c = t + i * 256;
    out[(size_t)blockIdx.x * 1024 + c] = (v[i] - mu) * rstd * gamma[c] + beta[c];
  }
}

// ------------------------------------------------------------------- launch
extern "C" void kernel_launch(void* const* d_in, const int* in_sizes, int n_in,
                              void* d_out, int out_size, void* d_ws, size_t ws_size,
                              hipStream_t stream) {
  const float* Q  = (const float*)d_in[0];
  const float* K  = (const float*)d_in[1];
  const float* V  = (const float*)d_in[2];
  const unsigned char* mask = (const unsigned char*)d_in[3];
  const float* Wq = (const float*)d_in[4];
  const float* bq = (const float*)d_in[5];
  const float* Wk = (const float*)d_in[6];
  const float* bk = (const float*)d_in[7];
  const float* Wv = (const float*)d_in[8];
  const float* bv = (const float*)d_in[9];
  const float* Wo = (const float*)d_in[10];
  const float* bo = (const float*)d_in[11];
  const float* gamma = (const float*)d_in[12];
  const float* beta  = (const float*)d_in[13];

  float* out  = (float*)d_out;                       // [B,L,D]
  float* attn = out + (size_t)BB * LL * D_MODEL;     // [B,H,L,L]

  char* w = (char*)d_ws;
  const size_t MB = 1ull << 20;
  unsigned short* Qb   = (unsigned short*)(w + 0 * MB);
  unsigned short* Kb   = (unsigned short*)(w + 16 * MB);
  unsigned short* Vb   = (unsigned short*)(w + 32 * MB);
  unsigned short* Wqb  = (unsigned short*)(w + 48 * MB);
  unsigned short* Wkb  = (unsigned short*)(w + 50 * MB);
  unsigned short* Wvb  = (unsigned short*)(w + 52 * MB);
  unsigned short* Wob  = (unsigned short*)(w + 54 * MB);
  unsigned short* qb   = (unsigned short*)(w + 56 * MB);
  unsigned short* kb   = (unsigned short*)(w + 72 * MB);
  unsigned short* vb   = (unsigned short*)(w + 88 * MB);
  unsigned short* ctxb = (unsigned short*)(w + 104 * MB);
  float*          y    = (float*)(w + 120 * MB);

  const int nX = BB * LL * D_MODEL;        // 8,388,608
  const int nW = D_MODEL * D_MODEL;        // 1,048,576
  dim3 blk(256);

  cast_f32_bf16<<<dim3(nX / 1024), blk, 0, stream>>>(Q,  Qb,  nX);
  cast_f32_bf16<<<dim3(nX / 1024), blk, 0, stream>>>(K,  Kb,  nX);
  cast_f32_bf16<<<dim3(nX / 1024), blk, 0, stream>>>(V,  Vb,  nX);
  cast_f32_bf16<<<dim3(nW / 1024), blk, 0, stream>>>(Wq, Wqb, nW);
  cast_f32_bf16<<<dim3(nW / 1024), blk, 0, stream>>>(Wk, Wkb, nW);
  cast_f32_bf16<<<dim3(nW / 1024), blk, 0, stream>>>(Wv, Wvb, nW);
  cast_f32_bf16<<<dim3(nW / 1024), blk, 0, stream>>>(Wo, Wob, nW);

  dim3 gProj(64, 16);  // 8192/128 x 1024/64
  proj_gemm<<<gProj, blk, 0, stream>>>(Qb, Wqb, bq, qb);
  proj_gemm<<<gProj, blk, 0, stream>>>(Kb, Wkb, bk, kb);
  proj_gemm<<<gProj, blk, 0, stream>>>(Vb, Wvb, bv, vb);

  dim3 gS(16, 32, BB * N_HEADS);  // 2048/128 x 2048/64 x 64
  scores_gemm<<<gS, blk, 0, stream>>>(qb, kb, mask, attn);

  softmax_rows<<<dim3(BB * N_HEADS * LL), blk, 0, stream>>>(attn);

  dim3 gC(16, 1, BB * N_HEADS);   // 2048/128, N=64 fits one block column
  context_gemm<<<gC, blk, 0, stream>>>(attn, vb, ctxb);

  dim3 gO(64, 16);
  outproj_gemm<<<gO, blk, 0, stream>>>(ctxb, Wob, bo, Q, y);

  layernorm_rows<<<dim3(BB * LL), blk, 0, stream>>>(y, gamma, beta, out);
}
